// HGCN_18537078850140
// MI455X (gfx1250) — matrix operand order repaired and checked by reading, not compiled, not run on hardware
//
#include <hip/hip_runtime.h>

// HGCN for MI455X (gfx1250, wave32, WMMA).
// Dominant op wp = W W^T (4096 x 18000) done as fp16 WMMA Gram; the fp16
// weight matrix (151MB) fits in the 192MB global L2, so tile re-reads are
// L2-resident. GEMM uses double-buffered LDS K-slabs filled with CDNA5
// GLOBAL_LOAD_ASYNC_TO_LDS_B128 (ASYNCcnt) when available, plus software-
// pipelined B fragments so ds_load latency overlaps WMMA issue.
// All reductions fixed-order (deterministic, no atomics).

#define NUC   8000
#define NIC   10000
#define NTOT  18000
#define GG    4096
#define NU_PAD 8192
#define N_PAD  18432   // 8192 (users, padded) + 10240 (items, padded); %64==0
#define EPSV  1e-8f
#define ALPHAV 0.8f

typedef __attribute__((ext_vector_type(8)))  _Float16 v8h;
typedef __attribute__((ext_vector_type(16))) _Float16 v16h;
typedef __attribute__((ext_vector_type(8)))  float    v8f;
typedef int i32x4 __attribute__((vector_size(16)));

#if defined(__AMDGCN__) && __has_builtin(__builtin_amdgcn_global_load_async_to_lds_b128)
#define HAVE_ASYNC_LDS 1
typedef __attribute__((address_space(1))) i32x4* gas_i32x4_p;
typedef __attribute__((address_space(3))) i32x4* las_i32x4_p;
#else
#define HAVE_ASYNC_LDS 0
#endif

__device__ __forceinline__ void wait_async0() {
#if defined(__AMDGCN__)
#if __has_builtin(__builtin_amdgcn_s_wait_asynccnt)
    __builtin_amdgcn_s_wait_asynccnt(0);
#else
    asm volatile("s_wait_asynccnt 0" ::: "memory");
#endif
#endif
}

// ---------------------------------------------------------------- phase A1:
// k = x@Wk.T+bk (kept in f32 scratch), v = x@Wv.T+bv written transposed into
// Vt (128 x N_PAD fp16): rows 0..63 = user V, rows 64..127 = item V.
__global__ __launch_bounds__(256) void kv_kernel(
    const float* __restrict__ X,
    const float* __restrict__ uWk, const float* __restrict__ ubk,
    const float* __restrict__ uWv, const float* __restrict__ ubv,
    const float* __restrict__ iWk, const float* __restrict__ ibk,
    const float* __restrict__ iWv, const float* __restrict__ ibv,
    float* __restrict__ Kbuf, _Float16* __restrict__ Vt)
{
    __shared__ float xs[4][64];
    const int tid  = threadIdx.x;
    const int ni   = tid >> 6;
    const int d    = tid & 63;
    const int node = blockIdx.x * 4 + ni;   // 8000 % 4 == 0: no set straddle
    xs[ni][d] = X[(size_t)node * 64 + d];
    __syncthreads();
    const bool item = node >= NUC;
    const float* Wk = item ? iWk : uWk;
    const float* Wv = item ? iWv : uWv;
    float k = (item ? ibk : ubk)[d];
    float v = (item ? ibv : ubv)[d];
#pragma unroll
    for (int e = 0; e < 64; ++e) {
        float x = xs[ni][e];
        k += x * Wk[d * 64 + e];
        v += x * Wv[d * 64 + e];
    }
    Kbuf[(size_t)node * 64 + d] = k;
    const int np = item ? (NU_PAD + (node - NUC)) : node;
    Vt[(size_t)(d + (item ? 64 : 0)) * N_PAD + np] = (_Float16)v;
}

// ---------------------------------------------------------------- phase A2:
// t = Ws @ K (1x64) per set, then r = Wq^T t, c0 = bq.t + bs  ->  s[n]=x.r+c0
__global__ __launch_bounds__(256) void tvec_kernel(
    const float* __restrict__ Kbuf,
    const float* __restrict__ uWs, const float* __restrict__ uWq,
    const float* __restrict__ ubq, const float* __restrict__ ubs,
    const float* __restrict__ iWs, const float* __restrict__ iWq,
    const float* __restrict__ ibq, const float* __restrict__ ibs,
    float* __restrict__ rbuf)
{
    __shared__ float red[4][64];
    __shared__ float tv[64];
    const int set = blockIdx.x, tid = threadIdx.x;
    const int d = tid & 63, part = tid >> 6;
    const float* Ws = set ? iWs : uWs;
    const float* Wq = set ? iWq : uWq;
    const float* bq = set ? ibq : ubq;
    const float* bs = set ? ibs : ubs;
    const int n0  = set ? NUC : 0;
    const int cnt = set ? NIC : NUC;
    float acc = 0.f;
    for (int n = part; n < cnt; n += 4)
        acc += Ws[n] * Kbuf[(size_t)(n0 + n) * 64 + d];
    red[part][d] = acc;
    __syncthreads();
    if (part == 0) tv[d] = red[0][d] + red[1][d] + red[2][d] + red[3][d];
    __syncthreads();
    if (tid < 64) {
        float r = 0.f;
        for (int dd = 0; dd < 64; ++dd) r += Wq[dd * 64 + tid] * tv[dd];
        rbuf[set * 65 + tid] = r;
    }
    if (tid == 64) {
        float c0 = bs[0];
        for (int dd = 0; dd < 64; ++dd) c0 += bq[dd] * tv[dd];
        rbuf[set * 65 + 64] = c0;
    }
}

// ---------------------------------------------------------------- phase A3:
// u[n] = exp(s[n]); s magnitudes are tiny (0.02-scale weights) -> exp safe
// without max-shift, mathematically equal to reference softmax.
__global__ __launch_bounds__(256) void uexp_kernel(
    const float* __restrict__ X, const float* __restrict__ rbuf,
    float* __restrict__ u)
{
    int n = blockIdx.x * 256 + threadIdx.x;
    if (n >= NTOT) return;
    const float* r = rbuf + (n >= NUC ? 65 : 0);
    float s = r[64];
    const float* x = X + (size_t)n * 64;
#pragma unroll
    for (int e = 0; e < 64; ++e) s += x[e] * r[e];
    u[n] = expf(s);
}

// ---------------------------------------------------------------- phase B:
// B[g, np] = H[n, g] * u[n]  (fp16, transposed via LDS tile; pads -> 0)
__global__ __launch_bounds__(256) void build_kernel(
    const int* __restrict__ H, const float* __restrict__ u,
    _Float16* __restrict__ B)
{
    __shared__ _Float16 tileT[64][66];
    const int tid = threadIdx.x;
    const int n0 = blockIdx.x * 64, g0 = blockIdx.y * 64;
    const int gl = tid & 63;
    const int q  = tid >> 6;
#pragma unroll
    for (int r = 0; r < 16; ++r) {
        int nl = q * 16 + r;
        int np = n0 + nl;
        int n; bool valid;
        if (np < NU_PAD) { n = np;                  valid = np < NUC; }
        else             { n = NUC + (np - NU_PAD); valid = np < NU_PAD + NIC; }
        float val = 0.f;
        if (valid && H[(size_t)n * GG + g0 + gl]) val = u[n];
        tileT[gl][nl] = (_Float16)val;
    }
    __syncthreads();
#pragma unroll
    for (int r = 0; r < 16; ++r) {
        int gg = q * 16 + r;
        B[(size_t)(g0 + gg) * N_PAD + n0 + gl] = tileT[gg][gl];
    }
}

// ---------------------------------------------------------------- phase Z:
// Z_u[g], Z_i[g] = per-set row sums of B (deterministic tree reduce).
__global__ __launch_bounds__(256) void zrow_kernel(
    const _Float16* __restrict__ B, float* __restrict__ Z)
{
    __shared__ float r0[256], r1[256];
    const int g = blockIdx.x, tid = threadIdx.x;
    const v8h* p = (const v8h*)(B + (size_t)g * N_PAD);
    float su = 0.f, si = 0.f;
    for (int c = tid; c < NU_PAD / 8; c += 256) {
        v8h v = p[c];
#pragma unroll
        for (int e = 0; e < 8; ++e) su += (float)v[e];
    }
    for (int c = NU_PAD / 8 + tid; c < N_PAD / 8; c += 256) {
        v8h v = p[c];
#pragma unroll
        for (int e = 0; e < 8; ++e) si += (float)v[e];
    }
    r0[tid] = su; r1[tid] = si;
    __syncthreads();
    for (int s = 128; s > 0; s >>= 1) {
        if (tid < s) { r0[tid] += r0[tid + s]; r1[tid] += r1[tid + s]; }
        __syncthreads();
    }
    if (tid == 0) { Z[g] = r0[0]; Z[GG + g] = r1[0]; }
}

// ---------------------------------------------------------------- phase C:
// B[g, n] /= Z_set(n)[g]   (in place; Z<=0 -> 0, degenerate empty column)
__global__ __launch_bounds__(256) void scale_kernel(
    _Float16* __restrict__ B, const float* __restrict__ Z)
{
    const int idx = blockIdx.x * 256 + threadIdx.x;   // over GG * (N_PAD/8)
    const int g = idx / (N_PAD / 8);
    const int c = idx % (N_PAD / 8);
    const int set = (c >= NU_PAD / 8);
    const float z = Z[set * GG + g];
    const float sc = (z > 0.f) ? 1.f / z : 0.f;
    v8h* p = (v8h*)B;
    v8h v = p[idx];
    v8h o;
#pragma unroll
    for (int e = 0; e < 8; ++e) o[e] = (_Float16)((float)v[e] * sc);
    p[idx] = o;
}

// ---------------------------------------------------------------- GEMM:
// C[i,j] = sum_k P[i,k] * Q[j,k]   (P,Q fp16 row-major, shared K dim).
// Used for wp = B B^T (grid 32x32) and gf = B Vt^T (grid 32x1).
// 8 waves/block, 128x128 macro-tile, wave w owns rows [16w,16w+16).
// Double-buffered LDS K-slabs of 64 (pitch 72 halves: conflict-free 16B lane
// reads) filled by async global->LDS loads; B fragments software-pipelined.
#define KSLAB 64
#define LDSP  72
__global__ __launch_bounds__(256) void gemm_kernel(
    const _Float16* __restrict__ P, const _Float16* __restrict__ Q,
    float* __restrict__ C, int ldc, int K)
{
    __shared__ _Float16 lp[2][128 * LDSP];
    __shared__ _Float16 lq[2][128 * LDSP];
    const int tid  = threadIdx.x;
    const int wave = tid >> 5, lane = tid & 31;
    const int half = lane >> 4, mrow = lane & 15;
    const int i0 = blockIdx.x * 128, j0 = blockIdx.y * 128;

    v8f acc[8] = {};

    auto stage = [&](int buf, int kb) {
#pragma unroll
        for (int r = 0; r < 4; ++r) {
            int c = tid + 256 * r;          // 1024 chunks of 16B per slab
            int row = c >> 3, seg = c & 7;
            const _Float16* gp = P + (size_t)(i0 + row) * K + kb + seg * 8;
            const _Float16* gq = Q + (size_t)(j0 + row) * K + kb + seg * 8;
            _Float16* dp = &lp[buf][row * LDSP + seg * 8];
            _Float16* dq = &lq[buf][row * LDSP + seg * 8];
#if HAVE_ASYNC_LDS
            __builtin_amdgcn_global_load_async_to_lds_b128(
                (gas_i32x4_p)gp, (las_i32x4_p)dp, 0, 0);
            __builtin_amdgcn_global_load_async_to_lds_b128(
                (gas_i32x4_p)gq, (las_i32x4_p)dq, 0, 0);
#else
            *(uint4*)dp = *(const uint4*)gp;
            *(uint4*)dq = *(const uint4*)gq;
            if (kb + KSLAB < K) {           // global_prefetch next K slab
                __builtin_prefetch(gp + KSLAB, 0, 1);
                __builtin_prefetch(gq + KSLAB, 0, 1);
            }
#endif
        }
    };

    stage(0, 0);
#if HAVE_ASYNC_LDS
    wait_async0();
#endif
    __syncthreads();

    int cur = 0;
    for (int kb = 0; kb < K; kb += KSLAB) {
        const int nxt = cur ^ 1;
        if (kb + KSLAB < K) stage(nxt, kb + KSLAB);   // overlap with compute

        const _Float16* lpb = lp[cur];
        const _Float16* lqb = lq[cur];
#pragma unroll
        for (int kk = 0; kk < KSLAB; kk += 32) {
            // A fragment (16x32 f16): lane L%16 = row M, K chunks per L/16
            const _Float16* ap = lpb + (wave * 16 + mrow) * LDSP + kk + half * 8;
            v8h a0 = *(const v8h*)(ap);
            v8h a1 = *(const v8h*)(ap + 16);
            v16h A = __builtin_shufflevector(a0, a1, 0, 1, 2, 3, 4, 5, 6, 7,
                                             8, 9, 10, 11, 12, 13, 14, 15);
            // B fragment (32x16 f16): lane L%16 = col N, K chunks per L/16.
            // Software pipeline: load frag jt+1 before issuing WMMA jt so the
            // backend can use partial dscnt waits instead of dscnt==0.
            const _Float16* bbase = lqb + mrow * LDSP + kk + half * 8;
            v8h b0 = *(const v8h*)(bbase);
            v8h b1 = *(const v8h*)(bbase + 16);
#pragma unroll
            for (int jt = 0; jt < 8; ++jt) {
                v8h n0 = b0, n1 = b1;
                if (jt < 7) {
                    const _Float16* bp = bbase + (jt + 1) * 16 * LDSP;
                    n0 = *(const v8h*)(bp);
                    n1 = *(const v8h*)(bp + 16);
                }
                v16h Bf = __builtin_shufflevector(b0, b1, 0, 1, 2, 3, 4, 5, 6, 7,
                                                  8, 9, 10, 11, 12, 13, 14, 15);
                acc[jt] = __builtin_amdgcn_wmma_f32_16x16x32_f16(
                    false, A, false, Bf, (short)0, acc[jt], false, false);
                b0 = n0; b1 = n1;
            }
        }
#if HAVE_ASYNC_LDS
        wait_async0();      // next buffer fully in LDS (per-wave ASYNCcnt)
#endif
        __syncthreads();    // all waves done reading cur + see nxt complete
        cur = nxt;
    }

    // C/D layout: VGPR v -> M = half*8 + v, N = lane%16
#pragma unroll
    for (int jt = 0; jt < 8; ++jt) {
        const int col = j0 + jt * 16 + mrow;
#pragma unroll
        for (int v = 0; v < 8; ++v) {
            const int row = i0 + wave * 16 + half * 8 + v;
            C[(size_t)row * ldc + col] = acc[jt][v];
        }
    }
}

// ---------------------------------------------------------------- norms:
// norms[g] = sqrt(wp[g,g])  (== ||W_g||2 exactly, diag of the Gram)
__global__ void norms_kernel(const float* __restrict__ wp,
                             float* __restrict__ norms)
{
    int g = blockIdx.x * 256 + threadIdx.x;
    if (g < GG) norms[g] = sqrtf(wp[(size_t)g * GG + g]);
}

// ---------------------------------------------------------------- final:
// A = wp/(n n^T + eps) (symmetric) -> msg = A gf, deg = rowsum(A),
// agg = a*gf + (1-a)*msg/deg, out = sigmoid(agg @ gW^T + gb)
__global__ __launch_bounds__(128) void final_kernel(
    const float* __restrict__ wp, const float* __restrict__ norms,
    const float* __restrict__ gf, const float* __restrict__ gW,
    const float* __restrict__ gb, float* __restrict__ out)
{
    __shared__ float sh[128];
    __shared__ float red[128];
    const int g = blockIdx.x, t = threadIdx.x;
    const float ng = norms[g];
    const float* wr = wp + (size_t)g * GG;
    float msg = 0.f, degp = 0.f;
    for (int h0 = 0; h0 < GG; h0 += 128) {
        float a = wr[h0 + t] / (ng * norms[h0 + t] + EPSV);
        sh[t] = a; degp += a;
        __syncthreads();
#pragma unroll 8
        for (int j = 0; j < 128; ++j) msg += sh[j] * gf[(size_t)(h0 + j) * 128 + t];
        __syncthreads();
    }
    red[t] = degp;
    __syncthreads();
    for (int s = 64; s > 0; s >>= 1) {
        if (t < s) red[t] += red[t + s];
        __syncthreads();
    }
    const float deg = red[0];
    const float hn = (deg > 0.f) ? msg / deg : 0.f;
    const float agg = ALPHAV * gf[(size_t)g * 128 + t] + (1.f - ALPHAV) * hn;
    __syncthreads();
    sh[t] = agg;
    __syncthreads();
    if (t < 64) {
        float s = gb[t];
#pragma unroll 8
        for (int e = 0; e < 128; ++e) s += sh[e] * gW[t * 128 + e];
        out[(size_t)g * 64 + t] = 1.f / (1.f + expf(-s));
    }
}

// ----------------------------------------------------------------
extern "C" void kernel_launch(void* const* d_in, const int* in_sizes, int n_in,
                              void* d_out, int out_size, void* d_ws, size_t ws_size,
                              hipStream_t stream)
{
    (void)in_sizes; (void)n_in; (void)out_size; (void)ws_size;
    const int*   H   = (const int*)  d_in[0];
    const float* X   = (const float*)d_in[1];
    // d_in[2] = num_groups (compile-time 4096)
    const float* uWq = (const float*)d_in[3];
    const float* ubq = (const float*)d_in[4];
    const float* uWk = (const float*)d_in[5];
    const float* ubk = (const float*)d_in[6];
    const float* uWv = (const float*)d_in[7];
    const float* ubv = (const float*)d_in[8];
    const float* uWs = (const float*)d_in[9];
    const float* ubs = (const float*)d_in[10];
    const float* iWq = (const float*)d_in[11];
    const float* ibq = (const float*)d_in[12];
    const float* iWk = (const float*)d_in[13];
    const float* ibk = (const float*)d_in[14];
    const float* iWv = (const float*)d_in[15];
    const float* ibv = (const float*)d_in[16];
    const float* iWs = (const float*)d_in[17];
    const float* ibs = (const float*)d_in[18];
    const float* gW  = (const float*)d_in[19];
    const float* gb  = (const float*)d_in[20];
    float* out = (float*)d_out;

    char* w = (char*)d_ws;
    size_t off = 0;
    auto alloc = [&](size_t bytes) -> void* {
        void* p = w + off;
        off += (bytes + 255) & ~(size_t)255;
        return p;
    };
    _Float16* B     = (_Float16*)alloc((size_t)GG * N_PAD * 2);   // 151 MB
    float*    wp    = (float*)   alloc((size_t)GG * GG * 4);      // 67 MB
    _Float16* Vt    = (_Float16*)alloc((size_t)128 * N_PAD * 2);  // 4.7 MB
    float*    Kbuf  = (float*)   alloc((size_t)NTOT * 64 * 4);
    float*    u     = (float*)   alloc((size_t)NTOT * 4);
    float*    gf    = (float*)   alloc((size_t)GG * 128 * 4);
    float*    Z     = (float*)   alloc((size_t)2 * GG * 4);
    float*    rbuf  = (float*)   alloc((size_t)2 * 65 * 4);
    float*    norms = (float*)   alloc((size_t)GG * 4);

    (void)hipMemsetAsync(Vt, 0, (size_t)128 * N_PAD * 2, stream);

    kv_kernel<<<NTOT / 4, 256, 0, stream>>>(X, uWk, ubk, uWv, ubv,
                                            iWk, ibk, iWv, ibv, Kbuf, Vt);
    tvec_kernel<<<2, 256, 0, stream>>>(Kbuf, uWs, uWq, ubq, ubs,
                                       iWs, iWq, ibq, ibs, rbuf);
    uexp_kernel<<<(NTOT + 255) / 256, 256, 0, stream>>>(X, rbuf, u);
    build_kernel<<<dim3(N_PAD / 64, GG / 64), 256, 0, stream>>>(H, u, B);
    zrow_kernel<<<GG, 256, 0, stream>>>(B, Z);
    scale_kernel<<<(GG * (N_PAD / 8)) / 256, 256, 0, stream>>>(B, Z);
    gemm_kernel<<<dim3(GG / 128, GG / 128), 256, 0, stream>>>(B, B, wp, GG, N_PAD);
    gemm_kernel<<<dim3(GG / 128, 1), 256, 0, stream>>>(B, Vt, gf, 128, N_PAD);
    norms_kernel<<<(GG + 255) / 256, 256, 0, stream>>>(wp, norms);
    final_kernel<<<GG, 128, 0, stream>>>(wp, norms, gf, gW, gb, out);
}
